// GraniteMoeHybridMambaLayer_56650618635045
// MI455X (gfx1250) — matrix operand order repaired and checked by reading, not compiled, not run on hardware
//
#include <hip/hip_runtime.h>
#include <math.h>

// ---------------- problem constants ----------------
#define HIDC 2048
#define NH   64          // heads
#define PDIM 64          // head dim
#define NSTATE 128       // state dim N
#define KCONV 4
#define CSZ  256         // chunk size
#define INTERC (2*HIDC)                    // 4096
#define CONVD  (INTERC + 2*NSTATE)         // 4352
#define PROJC  (INTERC + CONVD + NH)       // 8512
#define LSEQ 4096
#define NC   (LSEQ/CSZ)                    // 16
#define EPSF 1e-6f

typedef __bf16 bf16;
typedef __attribute__((ext_vector_type(16))) __bf16 v16bf;
typedef __attribute__((ext_vector_type(8)))  float  v8f;
typedef __attribute__((ext_vector_type(8)))  __bf16 bf16x8;

// ---------------- CDNA5 async copy (ASYNCcnt path) ----------------
// GLOBAL_LOAD_ASYNC_TO_LDS_B128: per-lane 16B global -> LDS, no VGPR data.
__device__ inline void async_ld_b128(void* lds_dst, const void* gsrc) {
  unsigned lds_off = (unsigned)(unsigned long long)lds_dst;   // flat shared addr, low 32b = LDS offset
  unsigned long long ga = (unsigned long long)gsrc;
  asm volatile("global_load_async_to_lds_b128 %0, %1, off"
               :: "v"(lds_off), "v"(ga) : "memory");
}
__device__ inline void wait_async0() {
  asm volatile("s_wait_asynccnt 0x0" ::: "memory");
}

// ---------------- WMMA helpers ----------------
__device__ inline v8f wmma_bf16(v16bf a, v16bf b, v8f c) {
  return __builtin_amdgcn_wmma_f32_16x16x32_bf16(false, a, false, b, (short)0, c, false, false);
}

// A fragment (16x32, 16-bit) from row-major S[m][k], leading dim ld.
__device__ inline v16bf frag_a_rm(const bf16* s, int row0, int k0, int ld) {
  int lane = threadIdx.x & 31;
  int m = row0 + (lane & 15);
  int hh = lane >> 4;
  v16bf a;
#pragma unroll
  for (int j = 0; j < 8; ++j) {
    int kb = (j < 4) ? (hh * 8 + 2 * j) : (16 + hh * 8 + 2 * (j - 4));
    a[2 * j]     = s[m * ld + k0 + kb];
    a[2 * j + 1] = s[m * ld + k0 + kb + 1];
  }
  return a;
}

// A fragment from k-major storage S[k][m] (i.e. A = S^T)
__device__ inline v16bf frag_a_km(const bf16* s, int row0, int k0, int ld) {
  int lane = threadIdx.x & 31;
  int m = row0 + (lane & 15);
  int hh = lane >> 4;
  v16bf a;
#pragma unroll
  for (int j = 0; j < 8; ++j) {
    int kb = (j < 4) ? (hh * 8 + 2 * j) : (16 + hh * 8 + 2 * (j - 4));
    a[2 * j]     = s[(k0 + kb) * ld + m];
    a[2 * j + 1] = s[(k0 + kb + 1) * ld + m];
  }
  return a;
}

// B fragment (32x16) where data is W[n][k] row-major (k contiguous)
__device__ inline v16bf frag_b_nk(const bf16* s, int col0, int k0, int ld) {
  int lane = threadIdx.x & 31;
  int n = col0 + (lane & 15);
  int hh = lane >> 4;
  v16bf b;
#pragma unroll
  for (int j = 0; j < 8; ++j) {
    int kb = (j < 4) ? (hh * 8 + 2 * j) : (16 + hh * 8 + 2 * (j - 4));
    b[2 * j]     = s[n * ld + k0 + kb];
    b[2 * j + 1] = s[n * ld + k0 + kb + 1];
  }
  return b;
}

// B fragment where data is S[k][n] row-major (n contiguous)
__device__ inline v16bf frag_b_kn(const bf16* s, int col0, int k0, int ld) {
  int lane = threadIdx.x & 31;
  int n = col0 + (lane & 15);
  int hh = lane >> 4;
  v16bf b;
#pragma unroll
  for (int j = 0; j < 8; ++j) {
    int kb = (j < 4) ? (hh * 8 + 2 * j) : (16 + hh * 8 + 2 * (j - 4));
    b[2 * j]     = s[(k0 + kb) * ld + n];
    b[2 * j + 1] = s[(k0 + kb + 1) * ld + n];
  }
  return b;
}

// ---------------- elementwise / setup kernels ----------------
__global__ __launch_bounds__(256) void f32_to_bf16_kernel(const float* __restrict__ src,
                                                          bf16* __restrict__ dst, size_t n) {
  size_t i = (size_t)blockIdx.x * 256 + threadIdx.x;
  if (i < n) dst[i] = (bf16)src[i];
}

// C[M,N] = A[M,K] * B[N,K]^T  (bf16 inputs, fp32 accumulate, fp32 or bf16 out)
template <bool OUT_BF16>
__global__ __launch_bounds__(256) void gemm_nt_kernel(const bf16* __restrict__ A,
                                                      const bf16* __restrict__ B,
                                                      void* __restrict__ Cout,
                                                      int M, int N, int K) {
  __shared__ bf16 sA[128][40];
  __shared__ bf16 sB[128][40];
  int tid = threadIdx.x;
  int lane = tid & 31;
  int wave = tid >> 5;
  int wm = wave & 3;   // 4 waves along M (32 rows each)
  int wn = wave >> 2;  // 2 waves along N (64 cols each)
  int m0 = blockIdx.y * 128;
  int n0 = blockIdx.x * 128;

  v8f acc[2][4];
#pragma unroll
  for (int i = 0; i < 2; ++i)
#pragma unroll
    for (int j = 0; j < 4; ++j) {
      v8f z = {0.f, 0.f, 0.f, 0.f, 0.f, 0.f, 0.f, 0.f};
      acc[i][j] = z;
    }

  for (int k0 = 0; k0 < K; k0 += 32) {
    __syncthreads();     // previous compute done before LDS overwrite
    // 128x32 bf16 tiles: 512 x (8 bf16) vectors, 256 threads -> 2 passes, async to LDS
#pragma unroll
    for (int it = 0; it < 2; ++it) {
      int i8 = tid + it * 256;
      int r = i8 >> 2;
      int c = (i8 & 3) * 8;
      async_ld_b128(&sA[r][c], A + (size_t)(m0 + r) * K + k0 + c);
      int rg = n0 + r;
      if (rg >= N) rg = N - 1;
      async_ld_b128(&sB[r][c], B + (size_t)rg * K + k0 + c);
    }
    if (k0 + 32 < K) {
      int r = tid >> 1;
      int rg = n0 + r; if (rg >= N) rg = N - 1;
      __builtin_prefetch(A + (size_t)(m0 + r) * K + k0 + 32, 0, 1);
      __builtin_prefetch(B + (size_t)rg * K + k0 + 32, 0, 1);
    }
    wait_async0();
    __syncthreads();

    v16bf a0 = frag_a_rm(&sA[0][0], 32 * wm, 0, 40);
    v16bf a1 = frag_a_rm(&sA[0][0], 32 * wm + 16, 0, 40);
#pragma unroll
    for (int j = 0; j < 4; ++j) {
      v16bf bfrag = frag_b_nk(&sB[0][0], 64 * wn + 16 * j, 0, 40);
      acc[0][j] = wmma_bf16(a0, bfrag, acc[0][j]);
      acc[1][j] = wmma_bf16(a1, bfrag, acc[1][j]);
    }
  }

  int rbase = m0 + 32 * wm + ((lane >> 4) * 8);
  int cbase = n0 + 64 * wn + (lane & 15);
#pragma unroll
  for (int i = 0; i < 2; ++i)
#pragma unroll
    for (int j = 0; j < 4; ++j) {
      int col = cbase + 16 * j;
      if (col < N) {
#pragma unroll
        for (int r = 0; r < 8; ++r) {
          int row = rbase + 16 * i + r;
          size_t off = (size_t)row * N + col;
          if (OUT_BF16) ((bf16*)Cout)[off] = (bf16)acc[i][j][r];
          else          ((float*)Cout)[off] = acc[i][j][r];
        }
      }
    }
}

// depthwise causal conv (K=4) + bias + SiLU over proj columns [INTER, INTER+CONVD)
__global__ __launch_bounds__(256) void conv_silu_kernel(const bf16* __restrict__ proj,
                                                        const float* __restrict__ conv_w,
                                                        const float* __restrict__ conv_b,
                                                        bf16* __restrict__ hBC) {
  int e = blockIdx.x * 256 + threadIdx.x;
  if (e >= LSEQ * CONVD) return;
  int l = e / CONVD;
  int ch = e - l * CONVD;
  float acc = conv_b[ch];
#pragma unroll
  for (int k = 0; k < KCONV; ++k) {
    int li = l - (KCONV - 1) + k;
    if (li >= 0)
      acc += (float)proj[(size_t)li * PROJC + INTERC + ch] * conv_w[ch * KCONV + k];
  }
  float s = acc / (1.f + __expf(-acc));
  hBC[(size_t)l * CONVD + ch] = (bf16)s;
}

// dt = softplus(proj[..., INTER+CONVD+h] + dt_bias)   -> [L][NH] fp32
__global__ __launch_bounds__(256) void dt_kernel(const bf16* __restrict__ proj,
                                                 const float* __restrict__ dt_bias,
                                                 float* __restrict__ dt) {
  int e = blockIdx.x * 256 + threadIdx.x;
  if (e >= LSEQ * NH) return;
  int l = e / NH;
  int h = e - l * NH;
  float v = (float)proj[(size_t)l * PROJC + INTERC + CONVD + h] + dt_bias[h];
  dt[e] = (v > 20.f) ? v : log1pf(__expf(v));
}

// per (chunk, head) inclusive cumsum of dt*A   -> acum[NH][LSEQ]
__global__ __launch_bounds__(256) void cumsum_kernel(const float* __restrict__ dt,
                                                     const float* __restrict__ A_log,
                                                     float* __restrict__ acum) {
  int c = blockIdx.x;
  int h = blockIdx.y;
  int s = threadIdx.x;
  __shared__ float buf[CSZ];
  float Ah = -__expf(A_log[h]);
  int l = c * CSZ + s;
  buf[s] = dt[l * NH + h] * Ah;
  __syncthreads();
  for (int off = 1; off < CSZ; off <<= 1) {
    float add = (s >= off) ? buf[s - off] : 0.f;
    __syncthreads();
    buf[s] += add;
    __syncthreads();
  }
  acum[(size_t)h * LSEQ + l] = buf[s];
}

// Y_diag: per (s_tile, chunk, head): CB = C.B^T, mask*decay*dt, then M @ x -> y
__global__ __launch_bounds__(256) void ydiag_kernel(const bf16* __restrict__ hBC,
                                                    const float* __restrict__ acum,
                                                    const float* __restrict__ dt,
                                                    float* __restrict__ y) {
  int st = blockIdx.x;   // 0..3 (64-row s tile)
  int c  = blockIdx.y;
  int h  = blockIdx.z;
  __shared__ bf16 sC[64][136];
  __shared__ bf16 sB[64][136];
  __shared__ bf16 sM[64][72];
  __shared__ bf16 sX[64][72];
  __shared__ float aS[64], aZ[64], dZ[64];

  int tid = threadIdx.x, lane = tid & 31, wave = tid >> 5;
  int wm = wave & 3, wn = wave >> 2;
  const int CBASE = INTERC + NSTATE;  // C segment
  const int BBASE = INTERC;           // B segment
  int s0g = c * CSZ + st * 64;

  for (int i8 = tid; i8 < 1024; i8 += 256) {  // 64x128 C tile (async)
    int r = i8 >> 4, col = (i8 & 15) * 8;
    async_ld_b128(&sC[r][col], hBC + (size_t)(s0g + r) * CONVD + CBASE + col);
  }
  if (tid < 64) aS[tid] = acum[(size_t)h * LSEQ + s0g + tid];

  v8f yacc[2];
  {
    v8f z = {0.f, 0.f, 0.f, 0.f, 0.f, 0.f, 0.f, 0.f};
    yacc[0] = z; yacc[1] = z;
  }

  for (int zi = 0; zi < 4; ++zi) {
    int z0g = c * CSZ + zi * 64;
    __syncthreads();
    for (int i8 = tid; i8 < 1024; i8 += 256) {  // 64x128 B tile (async)
      int r = i8 >> 4, col = (i8 & 15) * 8;
      async_ld_b128(&sB[r][col], hBC + (size_t)(z0g + r) * CONVD + BBASE + col);
    }
    for (int i8 = tid; i8 < 512; i8 += 256) {   // 64x64 x tile (async)
      int r = i8 >> 3, col = (i8 & 7) * 8;
      async_ld_b128(&sX[r][col], hBC + (size_t)(z0g + r) * CONVD + h * PDIM + col);
    }
    if (tid < 64) {
      aZ[tid] = acum[(size_t)h * LSEQ + z0g + tid];
      dZ[tid] = dt[(z0g + tid) * NH + h];
    }
    wait_async0();
    __syncthreads();

    // CB tile: wave computes rows [16*wm,+16) x cols [32*wn,+32), K = 128
    v8f cb[2];
    {
      v8f z = {0.f, 0.f, 0.f, 0.f, 0.f, 0.f, 0.f, 0.f};
      cb[0] = z; cb[1] = z;
    }
#pragma unroll
    for (int k0 = 0; k0 < 128; k0 += 32) {
      v16bf a  = frag_a_rm(&sC[0][0], 16 * wm, k0, 136);
      v16bf b0 = frag_b_nk(&sB[0][0], 32 * wn, k0, 136);
      v16bf b1 = frag_b_nk(&sB[0][0], 32 * wn + 16, k0, 136);
      cb[0] = wmma_bf16(a, b0, cb[0]);
      cb[1] = wmma_bf16(a, b1, cb[1]);
    }
    // mask + decay + dt, stash M as bf16
    int rl = 16 * wm + ((lane >> 4) * 8);
    int cl0 = 32 * wn + (lane & 15);
#pragma unroll
    for (int j = 0; j < 2; ++j)
#pragma unroll
      for (int r = 0; r < 8; ++r) {
        int sr = rl + r;
        int zc = cl0 + 16 * j;
        int sg = st * 64 + sr, zg = zi * 64 + zc;
        float v = 0.f;
        if (sg >= zg) v = cb[j][r] * __expf(aS[sr] - aZ[zc]) * dZ[zc];
        sM[sr][zc] = (bf16)v;
      }
    __syncthreads();
    // Y += M(64x64) @ x(64x64), K = 64 over z
#pragma unroll
    for (int k0 = 0; k0 < 64; k0 += 32) {
      v16bf a  = frag_a_rm(&sM[0][0], 16 * wm, k0, 72);
      v16bf b0 = frag_b_kn(&sX[0][0], 32 * wn, k0, 72);
      v16bf b1 = frag_b_kn(&sX[0][0], 32 * wn + 16, k0, 72);
      yacc[0] = wmma_bf16(a, b0, yacc[0]);
      yacc[1] = wmma_bf16(a, b1, yacc[1]);
    }
  }

  int rbase = 16 * wm + ((lane >> 4) * 8);
  int cbase = 32 * wn + (lane & 15);
#pragma unroll
  for (int j = 0; j < 2; ++j)
#pragma unroll
    for (int r = 0; r < 8; ++r) {
      int row = rbase + r, col = cbase + 16 * j;
      y[(size_t)(s0g + row) * INTERC + h * PDIM + col] = yacc[j][r];
    }
}

// states[c][h][p][n] = sum_s B[s][n] * exp(alast - acum[s]) * dt[s] * x[s][p]
__global__ __launch_bounds__(256) void states_kernel(const bf16* __restrict__ hBC,
                                                     const float* __restrict__ acum,
                                                     const float* __restrict__ dt,
                                                     float* __restrict__ states) {
  int c = blockIdx.x;
  int h = blockIdx.y;
  __shared__ bf16 sBw[64][136];
  __shared__ bf16 sX[64][72];
  __shared__ float wrow[64];
  int tid = threadIdx.x, lane = tid & 31, wave = tid >> 5;
  int wm = wave & 3, wn = wave >> 2;  // rows n: 32*wm, cols p: 32*wn
  const int BBASE = INTERC;

  v8f acc[2][2];
#pragma unroll
  for (int i = 0; i < 2; ++i)
#pragma unroll
    for (int j = 0; j < 2; ++j) {
      v8f z = {0.f, 0.f, 0.f, 0.f, 0.f, 0.f, 0.f, 0.f};
      acc[i][j] = z;
    }

  float alast = acum[(size_t)h * LSEQ + c * CSZ + CSZ - 1];
  for (int sc = 0; sc < 4; ++sc) {
    int s0g = c * CSZ + sc * 64;
    __syncthreads();
    if (tid < 64) {
      int l = s0g + tid;
      wrow[tid] = __expf(alast - acum[(size_t)h * LSEQ + l]) * dt[l * NH + h];
    }
    for (int i8 = tid; i8 < 512; i8 += 256) {  // x tile (async)
      int r = i8 >> 3, col = (i8 & 7) * 8;
      async_ld_b128(&sX[r][col], hBC + (size_t)(s0g + r) * CONVD + h * PDIM + col);
    }
    __syncthreads();   // wrow visible
    for (int idx = tid; idx < 64 * 128; idx += 256) {  // B scaled (register path)
      int r = idx >> 7, col = idx & 127;
      sBw[r][col] = (bf16)((float)hBC[(size_t)(s0g + r) * CONVD + BBASE + col] * wrow[r]);
    }
    wait_async0();
    __syncthreads();
#pragma unroll
    for (int k0 = 0; k0 < 64; k0 += 32) {
      v16bf a0 = frag_a_km(&sBw[0][0], 32 * wm, k0, 136);
      v16bf a1 = frag_a_km(&sBw[0][0], 32 * wm + 16, k0, 136);
      v16bf b0 = frag_b_kn(&sX[0][0], 32 * wn, k0, 72);
      v16bf b1 = frag_b_kn(&sX[0][0], 32 * wn + 16, k0, 72);
      acc[0][0] = wmma_bf16(a0, b0, acc[0][0]);
      acc[0][1] = wmma_bf16(a0, b1, acc[0][1]);
      acc[1][0] = wmma_bf16(a1, b0, acc[1][0]);
      acc[1][1] = wmma_bf16(a1, b1, acc[1][1]);
    }
  }
  // store transposed into [p][n]
  int nb = 32 * wm + ((lane >> 4) * 8);
  int pb = 32 * wn + (lane & 15);
#pragma unroll
  for (int i = 0; i < 2; ++i)
#pragma unroll
    for (int j = 0; j < 2; ++j)
#pragma unroll
      for (int r = 0; r < 8; ++r) {
        int n = nb + 16 * i + r;
        int p = pb + 16 * j;
        states[(((size_t)c * NH + h) * PDIM + p) * NSTATE + n] = acc[i][j][r];
      }
}

// sequential scan over chunks: prev[c] = carry; carry = carry*decay(c) + states[c]
__global__ __launch_bounds__(256) void scan_kernel(const float* __restrict__ states,
                                                   const float* __restrict__ acum,
                                                   float* __restrict__ prev) {
  int e = blockIdx.x * 256 + threadIdx.x;
  if (e >= NH * PDIM * NSTATE) return;
  int h = e / (PDIM * NSTATE);
  int rem = e - h * (PDIM * NSTATE);
  float carry = 0.f;
  for (int c = 0; c < NC; ++c) {
    size_t off = ((size_t)c * NH + h) * (PDIM * NSTATE) + rem;
    prev[off] = carry;
    float cd = __expf(acum[(size_t)h * LSEQ + c * CSZ + CSZ - 1]);
    carry = carry * cd + states[off];
  }
}

// y += (C * exp(acum)) @ prev^T + D*x
__global__ __launch_bounds__(256) void yoff_kernel(const bf16* __restrict__ hBC,
                                                   const float* __restrict__ acum,
                                                   const float* __restrict__ prev,
                                                   const float* __restrict__ Dvec,
                                                   float* __restrict__ y) {
  int st = blockIdx.x;
  int c = blockIdx.y;
  int h = blockIdx.z;
  __shared__ bf16 sA[64][136];
  __shared__ bf16 sP[64][136];
  __shared__ float erow[64];
  int tid = threadIdx.x, lane = tid & 31, wave = tid >> 5;
  int wm = wave & 3, wn = wave >> 2;
  const int CBASE = INTERC + NSTATE;
  int s0g = c * CSZ + st * 64;

  if (tid < 64) erow[tid] = __expf(acum[(size_t)h * LSEQ + s0g + tid]);
  for (int idx = tid; idx < 64 * 128; idx += 256) {  // prev[p][n] -> bf16
    int r = idx >> 7, col = idx & 127;
    sP[r][col] = (bf16)prev[(((size_t)c * NH + h) * PDIM + r) * NSTATE + col];
  }
  __syncthreads();
  for (int idx = tid; idx < 64 * 128; idx += 256) {  // C scaled (needs erow)
    int r = idx >> 7, col = idx & 127;
    sA[r][col] = (bf16)((float)hBC[(size_t)(s0g + r) * CONVD + CBASE + col] * erow[r]);
  }
  __syncthreads();

  v8f acc[2];
  {
    v8f z = {0.f, 0.f, 0.f, 0.f, 0.f, 0.f, 0.f, 0.f};
    acc[0] = z; acc[1] = z;
  }
#pragma unroll
  for (int k0 = 0; k0 < 128; k0 += 32) {
    v16bf a  = frag_a_rm(&sA[0][0], 16 * wm, k0, 136);
    v16bf b0 = frag_b_nk(&sP[0][0], 32 * wn, k0, 136);
    v16bf b1 = frag_b_nk(&sP[0][0], 32 * wn + 16, k0, 136);
    acc[0] = wmma_bf16(a, b0, acc[0]);
    acc[1] = wmma_bf16(a, b1, acc[1]);
  }
  float Dh = Dvec[h];
  int rbase = 16 * wm + ((lane >> 4) * 8);
  int cbase = 32 * wn + (lane & 15);
#pragma unroll
  for (int j = 0; j < 2; ++j)
#pragma unroll
    for (int r = 0; r < 8; ++r) {
      int row = rbase + r, col = cbase + 16 * j;
      size_t off = (size_t)(s0g + row) * INTERC + h * PDIM + col;
      float xv = (float)hBC[(size_t)(s0g + row) * CONVD + h * PDIM + col];
      y[off] = y[off] + acc[j][r] + Dh * xv;
    }
}

// g = y * silu(gate); rmsnorm; * norm_w; -> bf16 for final GEMM
__global__ __launch_bounds__(256) void gate_norm_kernel(const float* __restrict__ y,
                                                        const bf16* __restrict__ proj,
                                                        const float* __restrict__ norm_w,
                                                        bf16* __restrict__ gn) {
  int l = blockIdx.x;
  int tid = threadIdx.x;
  __shared__ float red[256];
  float ss = 0.f;
  for (int j = tid; j < INTERC; j += 256) {
    float gate = (float)proj[(size_t)l * PROJC + j];
    float g = y[(size_t)l * INTERC + j] * (gate / (1.f + __expf(-gate)));
    ss += g * g;
  }
  red[tid] = ss;
  __syncthreads();
  for (int off = 128; off > 0; off >>= 1) {
    if (tid < off) red[tid] += red[tid + off];
    __syncthreads();
  }
  float inv = rsqrtf(red[0] / (float)INTERC + EPSF);
  for (int j = tid; j < INTERC; j += 256) {
    float gate = (float)proj[(size_t)l * PROJC + j];
    float g = y[(size_t)l * INTERC + j] * (gate / (1.f + __expf(-gate)));
    gn[(size_t)l * INTERC + j] = (bf16)(g * inv * norm_w[j]);
  }
}

// ---------------- launch ----------------
extern "C" void kernel_launch(void* const* d_in, const int* in_sizes, int n_in,
                              void* d_out, int out_size, void* d_ws, size_t ws_size,
                              hipStream_t stream) {
  (void)in_sizes; (void)n_in; (void)out_size; (void)ws_size;
  const float* hs      = (const float*)d_in[0];
  const float* w_in    = (const float*)d_in[1];
  const float* conv_w  = (const float*)d_in[2];
  const float* conv_b  = (const float*)d_in[3];
  const float* A_log   = (const float*)d_in[4];
  const float* Dvec    = (const float*)d_in[5];
  const float* dt_bias = (const float*)d_in[6];
  const float* norm_w  = (const float*)d_in[7];
  const float* w_out   = (const float*)d_in[8];
  float* out = (float*)d_out;

  char* p = (char*)d_ws;
  auto alloc = [&](size_t bytes) {
    char* r = p;
    p += (bytes + 255) & ~(size_t)255;
    return r;
  };
  bf16*  hs_b    = (bf16*)alloc((size_t)LSEQ * HIDC * 2);
  bf16*  win_b   = (bf16*)alloc((size_t)PROJC * HIDC * 2);
  bf16*  wout_b  = (bf16*)alloc((size_t)HIDC * INTERC * 2);
  bf16*  proj_b  = (bf16*)alloc((size_t)LSEQ * PROJC * 2);
  bf16*  hbc_b   = (bf16*)alloc((size_t)LSEQ * CONVD * 2);
  float* dt_f    = (float*)alloc((size_t)LSEQ * NH * 4);
  float* acum    = (float*)alloc((size_t)NH * LSEQ * 4);
  float* states  = (float*)alloc((size_t)NC * NH * PDIM * NSTATE * 4);
  float* prev    = (float*)alloc((size_t)NC * NH * PDIM * NSTATE * 4);
  float* yb      = (float*)alloc((size_t)LSEQ * INTERC * 4);
  bf16*  gn      = (bf16*)alloc((size_t)LSEQ * INTERC * 2);

  auto cdiv = [](size_t a, size_t b) { return (unsigned)((a + b - 1) / b); };

  f32_to_bf16_kernel<<<cdiv((size_t)LSEQ * HIDC, 256), 256, 0, stream>>>(hs, hs_b, (size_t)LSEQ * HIDC);
  f32_to_bf16_kernel<<<cdiv((size_t)PROJC * HIDC, 256), 256, 0, stream>>>(w_in, win_b, (size_t)PROJC * HIDC);
  f32_to_bf16_kernel<<<cdiv((size_t)HIDC * INTERC, 256), 256, 0, stream>>>(w_out, wout_b, (size_t)HIDC * INTERC);

  // in_proj: [4096, 8512] = hs[4096,2048] @ w_in[8512,2048]^T
  gemm_nt_kernel<true><<<dim3(cdiv(PROJC, 128), LSEQ / 128), 256, 0, stream>>>(
      hs_b, win_b, proj_b, LSEQ, PROJC, HIDC);

  conv_silu_kernel<<<cdiv((size_t)LSEQ * CONVD, 256), 256, 0, stream>>>(proj_b, conv_w, conv_b, hbc_b);
  dt_kernel<<<cdiv((size_t)LSEQ * NH, 256), 256, 0, stream>>>(proj_b, dt_bias, dt_f);
  cumsum_kernel<<<dim3(NC, NH), 256, 0, stream>>>(dt_f, A_log, acum);

  ydiag_kernel<<<dim3(4, NC, NH), 256, 0, stream>>>(hbc_b, acum, dt_f, yb);
  states_kernel<<<dim3(NC, NH), 256, 0, stream>>>(hbc_b, acum, dt_f, states);
  scan_kernel<<<cdiv((size_t)NH * PDIM * NSTATE, 256), 256, 0, stream>>>(states, acum, prev);
  yoff_kernel<<<dim3(4, NC, NH), 256, 0, stream>>>(hbc_b, acum, prev, Dvec, yb);

  gate_norm_kernel<<<LSEQ, 256, 0, stream>>>(yb, proj_b, norm_w, gn);

  // out_proj: [4096, 2048] = gn[4096,4096] @ w_out[2048,4096]^T
  gemm_nt_kernel<false><<<dim3(HIDC / 128, LSEQ / 128), 256, 0, stream>>>(
      gn, wout_b, out, LSEQ, HIDC, INTERC);
}
